// VQVAE_8392366096866
// MI455X (gfx1250) — matrix operand order repaired and checked by reading, not compiled, not run on hardware
//
#include <hip/hip_runtime.h>
#include <cstdint>
#include <cstddef>

// ---------------------------------------------------------------------------
// CDNA5 (gfx1250) VQ-VAE forward.  All heavy conv layers + the VQ code search
// run as implicit-GEMM on v_wmma_f32_16x16x32_f16 (f16 inputs, f32 accum).
// Blocks compute NP*16 pixels; K is chunked through LDS (KC) so each A
// fragment (weights) is reused for NP WMMAs.
// ---------------------------------------------------------------------------

typedef _Float16 v16h __attribute__((ext_vector_type(16)));
typedef _Float16 v8h  __attribute__((ext_vector_type(8)));
typedef float    v8f  __attribute__((ext_vector_type(8)));

__device__ __forceinline__ v16h join8(v8h lo, v8h hi) {
    v16h r;
#pragma unroll
    for (int i = 0; i < 8; ++i) { r[i] = lo[i]; r[i + 8] = hi[i]; }
    return r;
}

#define WMMA16x16x32(A, B, C) \
    __builtin_amdgcn_wmma_f32_16x16x32_f16(false, (A), false, (B), (short)0, (C), false, false)

// ---------------------------------------------------------------------------
// Small helper kernels
// ---------------------------------------------------------------------------
__global__ void cast_f2h(const float* __restrict__ a, _Float16* __restrict__ b, int n) {
    int i = blockIdx.x * blockDim.x + threadIdx.x;
    if (i < n) b[i] = (_Float16)a[i];
}

// 0.5*||e||^2 per codebook entry
__global__ void emb_norm(const float* __restrict__ emb, float* __restrict__ en) {
    int e = blockIdx.x * blockDim.x + threadIdx.x;
    if (e >= 512) return;
    float s = 0.f;
    for (int k = 0; k < 128; ++k) { float v = emb[e * 128 + k]; s += v * v; }
    en[e] = 0.5f * s;
}

// d1 (ConvTranspose 128->64, k4 s2 p1) weights re-laid out per output parity:
// Wt2[(ph*2+pw)][oc][ic*4 + a*2 + b] = d1_w[ic][oc][3-(ph+2a)][3-(pw+2b)]
__global__ void prep_wt_d1(const float* __restrict__ w, _Float16* __restrict__ out) {
    int t = blockIdx.x * blockDim.x + threadIdx.x;      // 4*64*512
    if (t >= 4 * 64 * 512) return;
    int k  = t & 511;
    int oc = (t >> 9) & 63;
    int pq = t >> 15;
    int ph = pq >> 1, pw = pq & 1;
    int ic = k >> 2, a = (k >> 1) & 1, b = k & 1;
    int kh = ph + 2 * a, kw = pw + 2 * b;
    out[t] = (_Float16)w[ic * 1024 + oc * 16 + (3 - kh) * 4 + (3 - kw)];
}

// e0: 3->64, k4 s2 p1, 256->128, fused ReLU, fp32 in -> f16 out (NCHW)
__global__ void conv_e0(const float* __restrict__ x, const float* __restrict__ w,
                        const float* __restrict__ b, _Float16* __restrict__ out) {
    size_t o = (size_t)blockIdx.x * blockDim.x + threadIdx.x;
    if (o >= (size_t)32 * 64 * 128 * 128) return;
    int ow = (int)(o & 127), oh = (int)((o >> 7) & 127);
    int oc = (int)((o >> 14) & 63), n = (int)(o >> 20);
    float s = b[oc];
    for (int ic = 0; ic < 3; ++ic)
        for (int kh = 0; kh < 4; ++kh) {
            int ih = oh * 2 + kh - 1;
            if ((unsigned)ih >= 256u) continue;
            const float* xr = x + (((size_t)n * 3 + ic) << 16) + ((size_t)ih << 8);
            const float* wr = w + oc * 48 + ic * 16 + kh * 4;
            for (int kw = 0; kw < 4; ++kw) {
                int iw = ow * 2 + kw - 1;
                if ((unsigned)iw >= 256u) continue;
                s += wr[kw] * xr[iw];
            }
        }
    out[o] = (_Float16)fmaxf(s, 0.f);
}

// d2: ConvTranspose 64->3, k4 s2 p1, 128->256, fp32 out (no activation)
__global__ void convT_d2(const _Float16* __restrict__ h, const float* __restrict__ w,
                         const float* __restrict__ b, float* __restrict__ out) {
    size_t o = (size_t)blockIdx.x * blockDim.x + threadIdx.x;
    if (o >= (size_t)32 * 3 * 256 * 256) return;
    int ow = (int)(o & 255), oh = (int)((o >> 8) & 255);
    int t2 = (int)(o >> 16);
    int oc = t2 % 3, n = t2 / 3;
    int ph = oh & 1, pw = ow & 1;
    int ihb = (oh + ph - 2) / 2;
    int iwb = (ow + pw - 2) / 2;
    float s = b[oc];
    for (int ic = 0; ic < 64; ++ic) {
        const _Float16* xr = h + (((size_t)n * 64 + ic) << 14);
        const float* wr = w + ic * 48 + oc * 16;
#pragma unroll
        for (int a = 0; a < 2; ++a) {
            int ih = ihb + a;
            if ((unsigned)ih >= 128u) continue;
            int kh = ph + 2 * a;
#pragma unroll
            for (int bb = 0; bb < 2; ++bb) {
                int iw = iwb + bb;
                if ((unsigned)iw >= 128u) continue;
                int kw = pw + 2 * bb;
                s += wr[(3 - kh) * 4 + (3 - kw)] * (float)xr[(ih << 7) + iw];
            }
        }
    }
    out[o] = s;
}

// ---------------------------------------------------------------------------
// Implicit-GEMM conv via WMMA.
// MODE 0: 3x3 s1 p1, Cin=128, 64x64      (KT=1152)
// MODE 1: 1x1,        Cin=128, 64x64      (KT=128)
// MODE 2: 4x4 s2 p1,  Cin=64,  128->64    (KT=1024)
// MODE 3: ConvT 4x4 s2 p1, Cin=128, 64->128, OC=64, parity-decomposed (KT=512)
// Block = MT waves; wave w owns OC tile w; block covers NP*16 pixels.
// K is staged through LDS in chunks of KC; each A fragment feeds NP WMMAs.
// ---------------------------------------------------------------------------
template <int MODE, int KT, int KC, int NP, int MT,
          bool PRERELU, bool POSTRELU, bool RESID, bool WRF32>
__global__ __launch_bounds__(MT * 32) void conv_gemm(
    const _Float16* __restrict__ Wh, const _Float16* __restrict__ X,
    const float* __restrict__ bias, const _Float16* __restrict__ resid,
    _Float16* __restrict__ outh, float* __restrict__ outf)
{
    static_assert(KT % KC == 0 && KC % 32 == 0, "bad K chunking");
    __shared__ __align__(16) _Float16 Bs[NP * 16][KC + 8];

    const int t = blockIdx.x;
    int n, oh, ow0, ph = 0, pw = 0, iwb = 0;
    if (MODE == 3) {
        // per parity-row: 64 columns -> 64/(16*NP) tiles of 16*NP
        constexpr int TPR = 64 / (16 * NP);
        const int j = t % TPR;
        pw  = (t / TPR) & 1;
        oh  = (t / (TPR * 2)) & 127;
        n   = t / (TPR * 2 * 128);
        ph  = oh & 1;
        ow0 = pw + 2 * (16 * NP) * j;     // ow = ow0 + 2*p
        iwb = pw + (16 * NP) * j - 1;     // iw = iwb + p + b
    } else {
        const int p0 = t * (16 * NP);
        n   = p0 >> 12;
        oh  = (p0 >> 6) & 63;
        ow0 = p0 & 63;
    }
    const int ihb = (MODE == 3) ? ((oh + ph - 2) / 2) : 0;

    const int lane = threadIdx.x & 31;
    const int wv   = threadIdx.x >> 5;
    const int l15  = lane & 15;
    const int hi   = lane >> 4;       // 0 | 1
    const int koff = hi << 3;         // A frag: K base offset (+0 / +8)
    const int kb   = hi << 3;         // B frag: K base offset (+0 / +8)

    const _Float16* Wb   = (MODE == 3) ? (Wh + (((size_t)(ph * 2 + pw) * 64) << 9)) : Wh;
    const _Float16* wrow = Wb + (size_t)(wv * 16 + l15) * KT;

    v8f acc[NP];
#pragma unroll
    for (int s = 0; s < NP; ++s) acc[s] = {};

    for (int kc0 = 0; kc0 < KT; kc0 += KC) {
        if (kc0) __syncthreads();   // previous chunk fully consumed

        // --- cooperative im2col gather into LDS (fused pre-ReLU) ---
        for (int i = threadIdx.x; i < NP * 16 * KC; i += MT * 32) {
            const int p = i & (NP * 16 - 1);          // pixel in block
            const int k = kc0 + (i / (NP * 16));
            float v = 0.f;
            if (MODE == 0) {
                const int ic = k / 9, r = k - ic * 9, kh = r / 3, kw = r - kh * 3;
                const int ih = oh + kh - 1, iw = ow0 + p + kw - 1;
                if ((unsigned)ih < 64u && (unsigned)iw < 64u)
                    v = (float)X[(((size_t)n * 128 + ic) << 12) + (ih << 6) + iw];
            } else if (MODE == 1) {
                v = (float)X[(((size_t)n * 128 + k) << 12) + (oh << 6) + ow0 + p];
            } else if (MODE == 2) {
                const int ic = k >> 4, r = k & 15, kh = r >> 2, kw = r & 3;
                const int ih = oh * 2 + kh - 1, iw = (ow0 + p) * 2 + kw - 1;
                if ((unsigned)ih < 128u && (unsigned)iw < 128u)
                    v = (float)X[(((size_t)n * 64 + ic) << 14) + (ih << 7) + iw];
            } else {
                const int ic = k >> 2, a = (k >> 1) & 1, b = k & 1;
                const int ih = ihb + a, iw = iwb + p + b;
                if ((unsigned)ih < 64u && (unsigned)iw < 64u)
                    v = (float)X[(((size_t)n * 128 + ic) << 12) + (ih << 6) + iw];
            }
            if (PRERELU) v = fmaxf(v, 0.f);
            Bs[p][k - kc0] = (_Float16)v;
        }
        __syncthreads();

        // --- WMMA over this K chunk; A fragment reused for NP pixel tiles ---
        for (int k0 = 0; k0 < KC; k0 += 32) {
            const int gk = kc0 + k0;
            if (gk + 64 < KT) __builtin_prefetch(wrow + gk + 64, 0, 1);
            const v8h alo = *(const v8h*)(wrow + gk + koff);        // A: k+h (h=0..7)
            const v8h ahi = *(const v8h*)(wrow + gk + koff + 16);   // A: k+16+h
            const v16h A = join8(alo, ahi);
#pragma unroll
            for (int s = 0; s < NP; ++s) {
                const v8h blo = *(const v8h*)(&Bs[s * 16 + l15][k0 + kb]);
                const v8h bhi = *(const v8h*)(&Bs[s * 16 + l15][k0 + kb + 16]);
                acc[s] = WMMA16x16x32(A, join8(blo, bhi), acc[s]);
            }
        }
    }

    constexpr int OC = (MODE == 3) ? 64 : 128;
    constexpr int SH = (MODE == 3) ? 14 : 12;   // log2(OH*OW)
    constexpr int SW = (MODE == 3) ? 7  : 6;    // log2(OW)
    const int mhi = hi << 3;
#pragma unroll
    for (int s = 0; s < NP; ++s) {
        const int p  = s * 16 + l15;
        const int ow = (MODE == 3) ? (ow0 + 2 * p) : (ow0 + p);
#pragma unroll
        for (int r = 0; r < 8; ++r) {
            const int oc = wv * 16 + r + mhi;   // C/D: VGPR r -> M = r + 8*(lane>=16)
            float v = acc[s][r] + bias[oc];
            const size_t o = (((size_t)n * OC + oc) << SH) + (oh << SW) + ow;
            if (RESID) v += (float)resid[o];
            if (POSTRELU) v = fmaxf(v, 0.f);
            outh[o] = (_Float16)v;
            if (WRF32) outf[o] = v;
        }
    }
}

// ---------------------------------------------------------------------------
// VQ argmin via WMMA: maximize z.e - 0.5*||e||^2 over 512 codes.
// Block = 8 waves, 16 pixels; each wave scans 4 code-tiles of 16.
// Branchless argmax update (v_cmp + v_cndmask, no exec-mask branches).
// ---------------------------------------------------------------------------
__global__ __launch_bounds__(256) void vq_argmin(
    const _Float16* __restrict__ ze, const _Float16* __restrict__ embh,
    const float* __restrict__ embn, int* __restrict__ idx)
{
    __shared__ __align__(16) _Float16 Zs[16][136];
    __shared__ float bv_s[8][16];
    __shared__ int   bi_s[8][16];

    const int p0  = blockIdx.x << 4;
    const int n   = p0 >> 12;
    const int oh  = (p0 >> 6) & 63;
    const int ow0 = p0 & 63;

    for (int i = threadIdx.x; i < 16 * 128; i += 256) {
        const int jj = i & 15, k = i >> 4;
        Zs[jj][k] = ze[(((size_t)n * 128 + k) << 12) + (oh << 6) + ow0 + jj];
    }
    __syncthreads();

    const int lane = threadIdx.x & 31;
    const int wv   = threadIdx.x >> 5;
    const int l15  = lane & 15;
    const int hi   = lane >> 4;
    const int koff = hi << 3;
    const int kb   = hi << 3;
    const int mhi  = hi << 3;

    float bv = -3.0e38f;
    int   bi = 0x7fffffff;

    for (int q = 0; q < 4; ++q) {
        const int ct = wv * 4 + q;                       // code tile 0..31
        const _Float16* er = embh + ((size_t)(ct * 16 + l15)) * 128;
        v8f acc = {};
#pragma unroll
        for (int k0 = 0; k0 < 128; k0 += 32) {
            const v8h alo = *(const v8h*)(er + k0 + koff);
            const v8h ahi = *(const v8h*)(er + k0 + koff + 16);
            const v8h blo = *(const v8h*)(&Zs[l15][k0 + kb]);
            const v8h bhi = *(const v8h*)(&Zs[l15][k0 + kb + 16]);
            acc = WMMA16x16x32(join8(alo, ahi), join8(blo, bhi), acc);
        }
#pragma unroll
        for (int r = 0; r < 8; ++r) {
            const int code = ct * 16 + r + mhi;
            const float s = acc[r] - embn[code];
            const bool better = (s > bv) | ((s == bv) & (code < bi));
            bv = better ? s : bv;
            bi = better ? code : bi;
        }
    }
    // lanes l and l+16 hold the same pixel (N = lane & 15): merge halves
    {
        const float ov = __shfl_xor(bv, 16, 32);
        const int   oi = __shfl_xor(bi, 16, 32);
        const bool better = (ov > bv) | ((ov == bv) & (oi < bi));
        bv = better ? ov : bv;
        bi = better ? oi : bi;
    }
    if (lane < 16) { bv_s[wv][lane] = bv; bi_s[wv][lane] = bi; }
    __syncthreads();
    if (threadIdx.x < 16) {
        float best = bv_s[0][threadIdx.x];
        int   bidx = bi_s[0][threadIdx.x];
#pragma unroll
        for (int w = 1; w < 8; ++w) {
            const float v  = bv_s[w][threadIdx.x];
            const int   ii = bi_s[w][threadIdx.x];
            const bool better = (v > best) | ((v == best) & (ii < bidx));
            best = better ? v : best;
            bidx = better ? ii : bidx;
        }
        idx[p0 + threadIdx.x] = bidx;
    }
}

// Gather: zq fp32 to d_out + dec_in f16 to workspace (dec_in == zq forward)
__global__ void vq_gather(const int* __restrict__ idx, const float* __restrict__ emb,
                          float* __restrict__ zq, _Float16* __restrict__ decin) {
    int t = blockIdx.x * blockDim.x + threadIdx.x;     // 131072 * 8
    if (t >= 131072 * 8) return;
    const int pix = t >> 3, part = t & 7;
    const int n = pix >> 12, oh = (pix >> 6) & 63, ow = pix & 63;
    const int e = idx[pix];
#pragma unroll
    for (int i = 0; i < 16; ++i) {
        const int c = part * 16 + i;
        const float v = emb[e * 128 + c];
        const size_t o = (((size_t)n * 128 + c) << 12) + (oh << 6) + ow;
        zq[o] = v;
        decin[o] = (_Float16)v;
    }
}

// ---------------------------------------------------------------------------
// Host side
// ---------------------------------------------------------------------------
extern "C" void kernel_launch(void* const* d_in, const int* in_sizes, int n_in,
                              void* d_out, int out_size, void* d_ws, size_t ws_size,
                              hipStream_t stream) {
    (void)in_sizes; (void)n_in; (void)out_size; (void)ws_size;

    const float* x       = (const float*)d_in[0];
    const float* e0_w    = (const float*)d_in[1];
    const float* e0_b    = (const float*)d_in[2];
    const float* e1_w    = (const float*)d_in[3];
    const float* e1_b    = (const float*)d_in[4];
    const float* e2_w    = (const float*)d_in[5];
    const float* e2_b    = (const float*)d_in[6];
    const float* er1c1_w = (const float*)d_in[7];
    const float* er1c1_b = (const float*)d_in[8];
    const float* er1c2_w = (const float*)d_in[9];
    const float* er1c2_b = (const float*)d_in[10];
    const float* er2c1_w = (const float*)d_in[11];
    const float* er2c1_b = (const float*)d_in[12];
    const float* er2c2_w = (const float*)d_in[13];
    const float* er2c2_b = (const float*)d_in[14];
    const float* dr1c1_w = (const float*)d_in[15];
    const float* dr1c1_b = (const float*)d_in[16];
    const float* dr1c2_w = (const float*)d_in[17];
    const float* dr1c2_b = (const float*)d_in[18];
    const float* dr2c1_w = (const float*)d_in[19];
    const float* dr2c1_b = (const float*)d_in[20];
    const float* dr2c2_w = (const float*)d_in[21];
    const float* dr2c2_b = (const float*)d_in[22];
    const float* emb     = (const float*)d_in[23];
    const float* d0_w    = (const float*)d_in[24];
    const float* d0_b    = (const float*)d_in[25];
    const float* d1_w    = (const float*)d_in[26];
    const float* d1_b    = (const float*)d_in[27];
    const float* d2_w    = (const float*)d_in[28];
    const float* d2_b    = (const float*)d_in[29];

    float* ze_out = (float*)d_out;              // (32,128,64,64)
    float* zq_out = ze_out + 16777216;          // (32,128,64,64)
    float* xhat   = zq_out + 16777216;          // (32,3,256,256)

    // --- carve workspace ---
    char* wp = (char*)d_ws;
    auto carve = [&](size_t bytes) {
        char* r = wp;
        wp += (bytes + 255) & ~(size_t)255;
        return r;
    };
    _Float16* wh_e1    = (_Float16*)carve(131072 * 2);
    _Float16* wh_e2    = (_Float16*)carve(147456 * 2);
    _Float16* wh_r1c1  = (_Float16*)carve(147456 * 2);
    _Float16* wh_r1c2  = (_Float16*)carve(16384 * 2);
    _Float16* wh_r2c1  = (_Float16*)carve(147456 * 2);
    _Float16* wh_r2c2  = (_Float16*)carve(16384 * 2);
    _Float16* wh_d0    = (_Float16*)carve(147456 * 2);
    _Float16* wh_dr1c1 = (_Float16*)carve(147456 * 2);
    _Float16* wh_dr1c2 = (_Float16*)carve(16384 * 2);
    _Float16* wh_dr2c1 = (_Float16*)carve(147456 * 2);
    _Float16* wh_dr2c2 = (_Float16*)carve(16384 * 2);
    _Float16* wh_emb   = (_Float16*)carve(65536 * 2);
    _Float16* wt2_d1   = (_Float16*)carve(131072 * 2);
    float*    embn     = (float*)carve(512 * 4);
    int*      idxbuf   = (int*)carve(131072 * 4);
    _Float16* big0     = (_Float16*)carve((size_t)33554432 * 2);  // (32,64,128,128)
    _Float16* bufA     = (_Float16*)carve((size_t)16777216 * 2);  // (32,128,64,64)
    _Float16* bufB     = (_Float16*)carve((size_t)16777216 * 2);
    _Float16* bufC     = (_Float16*)carve((size_t)16777216 * 2);

    // --- weight preparation (all tiny) ---
    auto cast = [&](const float* s, _Float16* d, int n) {
        cast_f2h<<<(n + 255) / 256, 256, 0, stream>>>(s, d, n);
    };
    cast(e1_w, wh_e1, 131072);
    cast(e2_w, wh_e2, 147456);
    cast(er1c1_w, wh_r1c1, 147456);
    cast(er1c2_w, wh_r1c2, 16384);
    cast(er2c1_w, wh_r2c1, 147456);
    cast(er2c2_w, wh_r2c2, 16384);
    cast(d0_w, wh_d0, 147456);
    cast(dr1c1_w, wh_dr1c1, 147456);
    cast(dr1c2_w, wh_dr1c2, 16384);
    cast(dr2c1_w, wh_dr2c1, 147456);
    cast(dr2c2_w, wh_dr2c2, 16384);
    cast(emb, wh_emb, 65536);
    prep_wt_d1<<<512, 256, 0, stream>>>(d1_w, wt2_d1);
    emb_norm<<<2, 256, 0, stream>>>(emb, embn);

    // block counts: 131072 pixels / (16*NP=32) = 4096 for 64x64 layers
    // --- encoder ---
    conv_e0<<<131072, 256, 0, stream>>>(x, e0_w, e0_b, big0);
    // e1: 64->128 k4 s2, post-ReLU                 (KT=1024, KC=512, NP=2)
    conv_gemm<2, 1024, 512, 2, 8, false, true, false, false><<<4096, 256, 0, stream>>>(
        wh_e1, big0, e1_b, nullptr, bufA, nullptr);
    // e2: 3x3                                      (KT=1152, KC=384, NP=2)
    conv_gemm<0, 1152, 384, 2, 8, false, false, false, false><<<4096, 256, 0, stream>>>(
        wh_e2, bufA, e2_b, nullptr, bufB, nullptr);
    // er1: relu->3x3 ; relu->1x1 + residual
    conv_gemm<0, 1152, 384, 2, 8, true, false, false, false><<<4096, 256, 0, stream>>>(
        wh_r1c1, bufB, er1c1_b, nullptr, bufC, nullptr);
    conv_gemm<1, 128, 128, 2, 8, true, false, true, false><<<4096, 256, 0, stream>>>(
        wh_r1c2, bufC, er1c2_b, bufB, bufA, nullptr);
    // er2 (also writes ze fp32 to d_out)
    conv_gemm<0, 1152, 384, 2, 8, true, false, false, false><<<4096, 256, 0, stream>>>(
        wh_r2c1, bufA, er2c1_b, nullptr, bufC, nullptr);
    conv_gemm<1, 128, 128, 2, 8, true, false, true, true><<<4096, 256, 0, stream>>>(
        wh_r2c2, bufC, er2c2_b, bufA, bufB, ze_out);

    // --- vector quantization ---
    vq_argmin<<<8192, 256, 0, stream>>>(bufB, wh_emb, embn, idxbuf);
    vq_gather<<<(131072 * 8 + 255) / 256, 256, 0, stream>>>(idxbuf, emb, zq_out, bufA);

    // --- decoder ---
    conv_gemm<0, 1152, 384, 2, 8, false, false, false, false><<<4096, 256, 0, stream>>>(
        wh_d0, bufA, d0_b, nullptr, bufC, nullptr);
    conv_gemm<0, 1152, 384, 2, 8, true, false, false, false><<<4096, 256, 0, stream>>>(
        wh_dr1c1, bufC, dr1c1_b, nullptr, bufB, nullptr);
    conv_gemm<1, 128, 128, 2, 8, true, false, true, false><<<4096, 256, 0, stream>>>(
        wh_dr1c2, bufB, dr1c2_b, bufC, bufA, nullptr);
    conv_gemm<0, 1152, 384, 2, 8, true, false, false, false><<<4096, 256, 0, stream>>>(
        wh_dr2c1, bufA, dr2c1_b, nullptr, bufB, nullptr);
    conv_gemm<1, 128, 128, 2, 8, true, false, true, false><<<4096, 256, 0, stream>>>(
        wh_dr2c2, bufB, dr2c2_b, bufA, bufC, nullptr);
    // d1: ConvT 128->64, post-ReLU (parity tiles; KT=KC=512, NP=2 -> 16384 blocks)
    conv_gemm<3, 512, 512, 2, 4, false, true, false, false><<<16384, 128, 0, stream>>>(
        wt2_d1, bufC, d1_b, nullptr, big0, nullptr);
    // d2: ConvT 64->3 -> x_hat fp32
    convT_d2<<<(6291456 + 255) / 256, 256, 0, stream>>>(big0, d2_w, d2_b, xhat);
}
